// TaskAwareMoE_55284819034176
// MI455X (gfx1250) — matrix operand order repaired and checked by reading, not compiled, not run on hardware
//
#include <hip/hip_runtime.h>
#include <math.h>

// ---------------------------------------------------------------------------
// Task-aware MoE forward, MI455X (gfx1250) wave32.
// Compute-bound (~620 GFLOP, ~4100 FLOP/B vs HBM). bf16 WMMA 16x16x32 with
// f32 accumulate; one-time weight transpose->bf16 so B operands are single
// contiguous 32B K-runs; FC=128 chunking so each weight B-tile is loaded by
// exactly one wave per block and reused across 4 M-tiles (L2-BW friendly);
// TDM (tensor_load_to_lds) stages the X tile; hardware tanh gelu.
// ---------------------------------------------------------------------------

typedef __attribute__((ext_vector_type(4)))  float        v4f;
typedef __attribute__((ext_vector_type(8)))  float        v8f;
typedef __attribute__((ext_vector_type(8)))  __bf16       v8bf;
typedef __attribute__((ext_vector_type(16))) __bf16       v16bf;
typedef __attribute__((ext_vector_type(4)))  unsigned int v4u;
typedef __attribute__((ext_vector_type(4)))  int          v4i;
typedef __attribute__((ext_vector_type(8)))  int          v8i;

#define B_   8
#define N_   2048
#define D_   512
#define FF_  2048
#define E_   8
#define NTOK (B_ * N_)

#define MT   64     // tokens per workgroup tile
#define FC   128    // ff chunk width (8 N-tiles -> 1 wave per N-tile)
#define XSB  520    // X LDS row stride (bf16): 16B aligned, banks spread
#define HSB  136    // H LDS row stride (bf16): 16B aligned, banks spread
#define GS   9      // gates row: 8 expert gates + omega

#if __has_builtin(__builtin_amdgcn_tensor_load_to_lds) && \
    __has_builtin(__builtin_amdgcn_s_wait_tensorcnt)
#define USE_TDM 1
#else
#define USE_TDM 0
#endif

__device__ __forceinline__ unsigned short f32_to_bf16(float f) {
    unsigned u = __float_as_uint(f);
    unsigned r = (u + 0x7FFFu + ((u >> 16) & 1u)) >> 16;   // round-nearest-even
    return (unsigned short)r;
}

__device__ __forceinline__ float gelu_fast(float x) {
#if __has_builtin(__builtin_amdgcn_tanhf)
    float y = 0.7978845608028654f * x * fmaf(0.044715f * x, x, 1.0f);
    return 0.5f * x * (1.0f + __builtin_amdgcn_tanhf(y));
#elif __has_builtin(__builtin_amdgcn_tanh_f32)
    float y = 0.7978845608028654f * x * fmaf(0.044715f * x, x, 1.0f);
    return 0.5f * x * (1.0f + __builtin_amdgcn_tanh_f32(y));
#else
    // branchless erf, Abramowitz-Stegun 7.1.26 (|err| < 1.5e-7)
    float z  = fabsf(x) * 0.70710678118654752f;
    float t  = 1.0f / fmaf(0.3275911f, z, 1.0f);
    float p  = t * fmaf(t, fmaf(t, fmaf(t, fmaf(t, 1.061405429f, -1.453152027f),
                                        1.421413741f), -0.284496736f), 0.254829592f);
    float er = copysignf(1.0f - p * __expf(-z * z), x);
    return 0.5f * x * (1.0f + er);
#endif
}

// A operand (16-bit A 16x32, contiguous K): two 16B chunks, K{kh*16..kh*16+7} ++ K{kh*16+8..kh*16+15}
__device__ __forceinline__ v16bf ld_a_bf16(const unsigned short* p0,
                                           const unsigned short* p1) {
    v8bf lo = *(const v8bf*)p0;
    v8bf hi = *(const v8bf*)p1;
    return __builtin_shufflevector(lo, hi, 0, 1, 2, 3, 4, 5, 6, 7,
                                   8, 9, 10, 11, 12, 13, 14, 15);
}

// ---------------------------------------------------------------------------
// Weight prep: transpose f32 [batch][R][C] -> bf16 [batch][C][R] (tiled LDS)
// ---------------------------------------------------------------------------
__global__ void transpose_to_bf16(const float* __restrict__ src,
                                  unsigned short* __restrict__ dst,
                                  int R, int C) {
    __shared__ float tile[32][33];
    const int batch = blockIdx.z;
    src += (size_t)batch * R * C;
    dst += (size_t)batch * R * C;
    const int c0 = blockIdx.x * 32, r0 = blockIdx.y * 32;
    const int lc = threadIdx.x & 31, lr8 = threadIdx.x >> 5;
#pragma unroll
    for (int i = 0; i < 4; ++i) {
        int r = lr8 + i * 8;
        tile[r][lc] = src[(size_t)(r0 + r) * C + c0 + lc];
    }
    __syncthreads();
#pragma unroll
    for (int i = 0; i < 4; ++i) {
        int r = lr8 + i * 8;                      // = source column index
        dst[(size_t)(c0 + r) * R + r0 + lc] = f32_to_bf16(tile[lc][r]);
    }
}

// ---------------------------------------------------------------------------
// task_bias[t][e] = task_embed[t,:] . gw[D:2D, e] + gb[e]
// ---------------------------------------------------------------------------
__global__ void taskbias_kernel(const float* __restrict__ task_embed,
                                const float* __restrict__ gw,
                                const float* __restrict__ gb,
                                float* __restrict__ tb) {
    int l = threadIdx.x, t = l >> 3, e = l & 7;
    float acc = gb[e];
    for (int d = 0; d < D_; ++d)
        acc += task_embed[t * D_ + d] * gw[(size_t)(D_ + d) * E_ + e];
    tb[t * E_ + e] = acc;
}

// ---------------------------------------------------------------------------
// Gating: one wave32 per token; writes raw logits + gates(8)/omega.
// ---------------------------------------------------------------------------
__global__ void gating_kernel(const float* __restrict__ tokens,
                              const int*   __restrict__ task_ids,
                              const float* __restrict__ gw,
                              const float* __restrict__ tb,
                              float* __restrict__ logits_out,
                              float* __restrict__ gatesW) {
    const int wave = threadIdx.x >> 5, lane = threadIdx.x & 31;
    const int token = blockIdx.x * 8 + wave;
    const float* x = tokens + (size_t)token * D_;

    float acc[E_];
#pragma unroll
    for (int e = 0; e < E_; ++e) acc[e] = 0.0f;
    for (int d = lane; d < D_; d += 32) {
        float xv = x[d];
        const float* g = gw + (size_t)d * E_;
#pragma unroll
        for (int e = 0; e < E_; ++e) acc[e] += xv * g[e];
    }
#pragma unroll
    for (int e = 0; e < E_; ++e) {
#pragma unroll
        for (int s = 16; s > 0; s >>= 1) acc[e] += __shfl_xor(acc[e], s, 32);
    }
    if (lane == 0) {
        const int task = task_ids[token / N_];
        float lg[E_];
#pragma unroll
        for (int e = 0; e < E_; ++e) {
            lg[e] = acc[e] + tb[task * E_ + e];
            logits_out[(size_t)token * E_ + e] = lg[e];
        }
        int i1 = 0;
#pragma unroll
        for (int e = 1; e < E_; ++e) if (lg[e] > lg[i1]) i1 = e;
        int i2 = (i1 == 0) ? 1 : 0;
#pragma unroll
        for (int e = 0; e < E_; ++e) if (e != i1 && lg[e] > lg[i2]) i2 = e;
        float ev = __expf(lg[i2] - lg[i1]);
        float p1 = 1.0f / (1.0f + ev);
        float* gr = gatesW + (size_t)token * GS;
#pragma unroll
        for (int e = 0; e < E_; ++e) gr[e] = 0.0f;
        gr[i1] = p1;
        gr[i2] = ev * p1;
        gr[8]  = 1.0f - p1;   // omega
    }
}

// ---------------------------------------------------------------------------
// Fused dense-MoE FFN, bf16 WMMA 16x16x32, f32 accumulate.
// Block: 256 thr (8 waves), tile = 64 tokens x full D=512.
// ---------------------------------------------------------------------------
__global__ __launch_bounds__(256)
void moe_ffn_kernel(const float* __restrict__ tokens,
                    const unsigned short* __restrict__ w1t,   // [E][FF][D] bf16
                    const float* __restrict__ b1,
                    const unsigned short* __restrict__ w2t,   // [E][D][FF] bf16
                    const float* __restrict__ b2,
                    const unsigned short* __restrict__ uw1t,  // [FF][D] bf16
                    const float* __restrict__ ub1,
                    const unsigned short* __restrict__ uw2t,  // [D][FF] bf16
                    const float* __restrict__ ub2,
                    const float* __restrict__ gatesW,
                    float* __restrict__ out) {
    extern __shared__ char smem[];
    float*          Xf = (float*)smem;                   // MT x D f32 (TDM dest)
    unsigned short* Xl = (unsigned short*)(Xf + MT * D_);// MT x XSB bf16
    unsigned short* Hl = Xl + MT * XSB;                  // MT x HSB bf16
    float*          Gl = (float*)(Hl + MT * HSB);        // MT x GS

    const int tid  = threadIdx.x;
    const int wave = tid >> 5;
    const int lane = tid & 31;
    const int ln   = lane & 15;
    const int kh   = lane >> 4;            // half-wave selector (ISA K split)
    const size_t tBase = (size_t)blockIdx.x * MT;
    const float* Xg = tokens + tBase * D_; // 64x512 tile, contiguous

    // ---- stage X tile: global f32 -> LDS via Tensor Data Mover ----
#if USE_TDM
    if (wave == 0) {
        const unsigned long long ga = (unsigned long long)(const void*)Xg;
        const unsigned lds_off = (unsigned)(size_t)(void*)Xf;
        v4u g0;
        g0.x = 1u;                                   // count=1 (valid D#)
        g0.y = lds_off;                              // lds_addr (bytes)
        g0.z = (unsigned)ga;                         // global_addr[31:0]
        g0.w = ((unsigned)(ga >> 32) & 0x1FFFFFFu) | (2u << 30); // addr hi | type=2
        v8i g1;
        g1[0] = 0x00020000;                          // data_size=4B
        g1[1] = (int)(512u << 16);                   // tensor_dim0 = 512
        g1[2] = (int)(64u << 16);                    // tensor_dim1 = 64
        g1[3] = (int)(512u << 16);                   // tile_dim0 = 512
        g1[4] = 64;                                  // tile_dim1 = 64
        g1[5] = 512;                                 // tensor_dim0_stride = 512
        g1[6] = 0; g1[7] = 0;
        v4i gz4; gz4[0] = gz4[1] = gz4[2] = gz4[3] = 0;  // 2D: groups 2/3 unused
        v8i gz8;
#pragma unroll
        for (int i = 0; i < 8; ++i) gz8[i] = 0;
        // amdgpu-toolchain (clang-23) 6-arg form: (g0, g1, g2, g3, g4, cpol)
        __builtin_amdgcn_tensor_load_to_lds(g0, g1, gz4, gz4, gz8, 0);
        __builtin_amdgcn_s_wait_tensorcnt(0);
    }
    __syncthreads();
#else
    for (int i = tid * 4; i < MT * D_; i += 1024)
        *(v4f*)(Xf + i) = *(const v4f*)(Xg + i);
    __syncthreads();
#endif
    // convert f32 tile -> padded bf16 operand layout; stage gates
    for (int i = tid * 4; i < MT * D_; i += 1024) {
        int row = i >> 9, col = i & (D_ - 1);
        v4f v = *(const v4f*)(Xf + i);
        union { unsigned short s[4]; unsigned long long u; } pk;
        pk.s[0] = f32_to_bf16(v.x); pk.s[1] = f32_to_bf16(v.y);
        pk.s[2] = f32_to_bf16(v.z); pk.s[3] = f32_to_bf16(v.w);
        *(unsigned long long*)(Xl + row * XSB + col) = pk.u;
    }
    for (int i = tid; i < MT * GS; i += 256) Gl[i] = gatesW[tBase * GS + i];
    __syncthreads();

    // persistent f32 accumulators: 4 M-tiles x 4 N-tiles of 16x16
    v8f acc[4][4];
#pragma unroll
    for (int mt = 0; mt < 4; ++mt)
#pragma unroll
        for (int nt = 0; nt < 4; ++nt)
#pragma unroll
            for (int r = 0; r < 8; ++r) acc[mt][nt][r] = 0.0f;

    const int wcol  = wave * 16 + ln;   // stage-1: this wave's H column
    const int dbase = wave * 64;        // stage-2: this wave's D column slice

    for (int e = 0; e < 9; ++e) {       // 8 experts + universal (e==8)
        const unsigned short* W1e = (e < 8) ? (w1t + (size_t)e * FF_ * D_) : uw1t;
        const unsigned short* W2e = (e < 8) ? (w2t + (size_t)e * D_ * FF_) : uw2t;
        const float*          B1e = (e < 8) ? (b1 + (size_t)e * FF_)       : ub1;

        for (int ff0 = 0; ff0 < FF_; ff0 += FC) {
            // ---- stage 1: H[:, wave*16..] = gelu(X @ W1 + b1) * gate ----
            // one wave per N-tile: B loaded once, reused across 4 M-tiles
            v8f h[4];
#pragma unroll
            for (int mt = 0; mt < 4; ++mt)
#pragma unroll
                for (int r = 0; r < 8; ++r) h[mt][r] = 0.0f;

            const unsigned short* Wcol = W1e + (size_t)(ff0 + wcol) * D_ + kh * 16;
#pragma unroll 2
            for (int k0 = 0; k0 < D_; k0 += 32) {
                v16bf bv = *(const v16bf*)(Wcol + k0);
#pragma unroll
                for (int mt = 0; mt < 4; ++mt) {
                    const unsigned short* xp = Xl + (mt * 16 + ln) * XSB + k0 + kh * 16;
                    v16bf a = ld_a_bf16(xp, xp + 8);
                    h[mt] = __builtin_amdgcn_wmma_f32_16x16x32_bf16(
                        false, a, false, bv, (short)0, h[mt], false, false);
                }
            }
            const float bias = B1e[ff0 + wcol];
#pragma unroll
            for (int mt = 0; mt < 4; ++mt)
#pragma unroll
                for (int r = 0; r < 8; ++r) {
                    const int row = mt * 16 + r + kh * 8;
                    float v = gelu_fast(h[mt][r] + bias) * Gl[row * GS + e];
                    Hl[row * HSB + wcol] = f32_to_bf16(v);
                }
            __syncthreads();

            // ---- stage 2: acc += H @ W2[ff0:+128, dbase:+64] ----
#pragma unroll
            for (int kk0 = 0; kk0 < FC; kk0 += 32) {
                v16bf a2[4], bw[4];
#pragma unroll
                for (int mt = 0; mt < 4; ++mt) {
                    const unsigned short* hp = Hl + (mt * 16 + ln) * HSB + kk0 + kh * 16;
                    a2[mt] = ld_a_bf16(hp, hp + 8);
                }
#pragma unroll
                for (int nt = 0; nt < 4; ++nt) {
                    const int col = dbase + nt * 16 + ln;
                    bw[nt] = *(const v16bf*)(W2e + (size_t)col * FF_ + ff0 + kk0 + kh * 16);
                }
#pragma unroll
                for (int mt = 0; mt < 4; ++mt)
#pragma unroll
                    for (int nt = 0; nt < 4; ++nt)
                        acc[mt][nt] = __builtin_amdgcn_wmma_f32_16x16x32_bf16(
                            false, a2[mt], false, bw[nt], (short)0,
                            acc[mt][nt], false, false);
            }
            __syncthreads();   // protect H before next chunk overwrites it
        }
    }

    // ---- epilogue: + sum_e g_e*b2[e,:] + omega*ub2, store f32 ----
#pragma unroll
    for (int nt = 0; nt < 4; ++nt) {
        const int d = dbase + nt * 16 + ln;
        float b2c[8];
#pragma unroll
        for (int e = 0; e < 8; ++e) b2c[e] = b2[(size_t)e * D_ + d];
        const float ub2c = ub2[d];
#pragma unroll
        for (int mt = 0; mt < 4; ++mt) {
#pragma unroll
            for (int r = 0; r < 8; ++r) {
                const int trow = mt * 16 + r + kh * 8;
                const float* g = Gl + trow * GS;
                float s = acc[mt][nt][r];
#pragma unroll
                for (int e = 0; e < 8; ++e) s += g[e] * b2c[e];
                s += g[8] * ub2c;
                out[(tBase + trow) * D_ + d] = s;
            }
        }
    }
}

// ---------------------------------------------------------------------------
extern "C" void kernel_launch(void* const* d_in, const int* in_sizes, int n_in,
                              void* d_out, int out_size, void* d_ws, size_t ws_size,
                              hipStream_t stream) {
    const float* tokens     = (const float*)d_in[0];
    const int*   task_ids   = (const int*)  d_in[1];
    const float* task_embed = (const float*)d_in[2];
    const float* gw         = (const float*)d_in[3];
    const float* gb         = (const float*)d_in[4];
    const float* w1         = (const float*)d_in[5];
    const float* b1         = (const float*)d_in[6];
    const float* w2         = (const float*)d_in[7];
    const float* b2         = (const float*)d_in[8];
    const float* uw1        = (const float*)d_in[9];
    const float* ub1        = (const float*)d_in[10];
    const float* uw2        = (const float*)d_in[11];
    const float* ub2        = (const float*)d_in[12];

    float* out_main   = (float*)d_out;                      // [B,N,D]
    float* out_logits = (float*)d_out + (size_t)NTOK * D_;  // [B,N,E]

    // workspace layout (bytes): tb | gates | bf16 transposed weights (~38.9MB)
    char* ws = (char*)d_ws;
    float*          tb     = (float*)(ws);
    float*          gatesW = (float*)(ws + 1024);
    unsigned short* w1t    = (unsigned short*)(ws + 1024 + (size_t)NTOK * GS * 4 + 64);
    unsigned short* w2t    = w1t  + (size_t)E_ * FF_ * D_;
    unsigned short* uw1t   = w2t  + (size_t)E_ * D_ * FF_;
    unsigned short* uw2t   = uw1t + (size_t)FF_ * D_;

    // ---- weight prep: transpose + bf16 convert (one-time, ~150MB traffic) ----
    transpose_to_bf16<<<dim3(FF_ / 32, D_ / 32, E_), 256, 0, stream>>>(w1, w1t, D_, FF_);
    transpose_to_bf16<<<dim3(D_ / 32, FF_ / 32, E_), 256, 0, stream>>>(w2, w2t, FF_, D_);
    transpose_to_bf16<<<dim3(FF_ / 32, D_ / 32, 1), 256, 0, stream>>>(uw1, uw1t, D_, FF_);
    transpose_to_bf16<<<dim3(D_ / 32, FF_ / 32, 1), 256, 0, stream>>>(uw2, uw2t, FF_, D_);

    taskbias_kernel<<<1, 32, 0, stream>>>(task_embed, gw, gb, tb);
    gating_kernel<<<NTOK / 8, 256, 0, stream>>>(tokens, task_ids, gw, tb,
                                                out_logits, gatesW);

    const size_t shmem = (size_t)MT * D_ * 4          // Xf f32 (TDM dest)
                       + (size_t)MT * XSB * 2         // X bf16
                       + (size_t)MT * HSB * 2         // H bf16
                       + (size_t)MT * GS * 4;         // gates
    moe_ffn_kernel<<<NTOK / MT, 256, shmem, stream>>>(
        tokens, w1t, b1, w2t, b2, uw1t, ub1, uw2t, ub2, gatesW, out_main);
}
